// HGCNSheafBuilderLowRank_46926812676181
// MI455X (gfx1250) — matrix operand order repaired and checked by reading, not compiled, not run on hardware
//
#include <hip/hip_runtime.h>

typedef __attribute__((ext_vector_type(16))) _Float16 v16h;
typedef __attribute__((ext_vector_type(8)))  float    v8f;

#define DD     6
#define RANK_  2
#define FF     64
#define OUTCH  30          // 2*D*RANK + D
#define HROW_STRIDE 68     // dwords per h-row in LDS: 64 data + 4 pad (keeps 16B alignment, spreads banks)

// ---------------------------------------------------------------------------
// Kernel 0: pre-swizzle W (128 x 30, zero-padded to 32 cols) into wave32 WMMA
// B-fragment layout, and emit a zero-padded 32-entry bias vector.
// 8 fragments: ntile in {0,1} (N=0..15 / 16..31), kchunk in {0..3}
// (K = 32*c .. 32*c+31). Per fragment, lane l holds 16 f16:
//   n = ntile*16 + (l & 15),  k = c*32 + (l>>4)*16 + h   (h = 0..15)
// stored contiguously so the main kernel does two coalesced b128 loads/lane.
// ---------------------------------------------------------------------------
__global__ void wprep_kernel(const float* __restrict__ W, const float* __restrict__ bias,
                             _Float16* __restrict__ wfrag, float* __restrict__ biasPad)
{
    int t    = threadIdx.x;        // 0..255
    int f    = t >> 5;             // fragment 0..7  (ntile*4 + c)
    int lane = t & 31;
    int ntile = f >> 2;
    int c     = f & 3;
    int n     = ntile * 16 + (lane & 15);
    _Float16* dst = wfrag + ((size_t)f * 32 + lane) * 16;
#pragma unroll
    for (int h = 0; h < 16; ++h) {
        int k = c * 32 + (lane >> 4) * 16 + h;
        float v = (n < OUTCH) ? W[(size_t)k * OUTCH + n] : 0.0f;
        dst[h] = (_Float16)v;
    }
    if (t < 32) biasPad[t] = (t < OUTCH) ? bias[t] : 0.0f;
}

// ---------------------------------------------------------------------------
// Kernel 1: segment means over D=6, emitted as f16 (halves L2 gather traffic;
// xm_h + em_h = 15.4 MB -> fully L2 resident on MI455X's 192 MB L2).
// ---------------------------------------------------------------------------
__global__ void means_kernel(const float* __restrict__ x, const float* __restrict__ e,
                             _Float16* __restrict__ xmh, _Float16* __restrict__ emh,
                             int N, int E)
{
    long t = (long)blockIdx.x * blockDim.x + threadIdx.x;
    long totalX = (long)N * FF;
    long total  = totalX + (long)E * FF;
    if (t >= total) return;
    const float inv = 1.0f / (float)DD;
    if (t < totalX) {
        long n = t >> 6; int fidx = (int)(t & 63);
        const float* p = x + n * (DD * FF) + fidx;
        float s = 0.0f;
#pragma unroll
        for (int d = 0; d < DD; ++d) s += p[(size_t)d * FF];
        xmh[t] = (_Float16)(s * inv);
    } else {
        long u = t - totalX;
        long n = u >> 6; int fidx = (int)(u & 63);
        const float* p = e + n * (DD * FF) + fidx;
        float s = 0.0f;
#pragma unroll
        for (int d = 0; d < DD; ++d) s += p[(size_t)d * FF];
        emh[u] = (_Float16)(s * inv);
    }
}

// ---------------------------------------------------------------------------
// Kernel 2: fused gather + GEMM(WMMA f16) + sigmoid + low-rank expansion.
// 1 wave per 16 edges; 4 waves per block with private LDS regions.
// ---------------------------------------------------------------------------
__global__ __launch_bounds__(128)
void sheaf_main_kernel(const _Float16* __restrict__ xmh, const _Float16* __restrict__ emh,
                       const int* __restrict__ rowi, const int* __restrict__ coli,
                       const _Float16* __restrict__ wfrag, const float* __restrict__ biasPad,
                       float* __restrict__ out, int nnz, int ntiles)
{
    __shared__ unsigned int hbuf[4][16 * HROW_STRIDE];  // 16 rows x 128 f16 (padded) per wave
    __shared__ float        sheaf[4][16 * 32];          // 16 rows x 32 sigmoid outputs per wave

    const int lane = threadIdx.x & 31;
    const int wave = threadIdx.x >> 5;
    const int tile = blockIdx.x * 4 + wave;
    if (tile >= ntiles) return;                          // wave-uniform
    const int base = tile * 16;

    // ---- Phase 1: gather h = [xm[row] | em[col]] rows into LDS as f16 ------
    unsigned int* hb = hbuf[wave];
    for (int r = 0; r < 16; ++r) {
        int i = base + r; if (i >= nnz) i = nnz - 1;
        int ri = rowi[i];
        int ci = coli[i];
        // lane covers dword (2 f16) 'lane' of xm row and 'lane' of em row
        unsigned int xv = *(const unsigned int*)(xmh + (size_t)ri * FF + lane * 2);
        unsigned int ev = *(const unsigned int*)(emh + (size_t)ci * FF + lane * 2);
        hb[r * HROW_STRIDE + lane]      = xv;   // k = 0..63
        hb[r * HROW_STRIDE + 32 + lane] = ev;   // k = 64..127
    }
    asm volatile("s_wait_dscnt 0" ::: "memory");  // cross-lane LDS handoff

    // ---- Phase 2: (16x128 f16) x (128x32 f16) -> 16x32 f32 via 8 WMMAs -----
    const int m  = lane & 15;
    const int hi = lane >> 4;
    v8f acc0 = {};  // N = 0..15
    v8f acc1 = {};  // N = 16..31 (cols 30,31 are zero-padded W)
    const unsigned int* rowp = hb + m * HROW_STRIDE;
    const int4* wf = (const int4*)wfrag;
#pragma unroll
    for (int c = 0; c < 4; ++c) {
        // A fragment (contiguous, matches wfrag): lane<16 -> K {0..15}, lane>=16 -> {16..31}
        union { int4 q[2]; v16h h; } a;
        a.q[0] = *(const int4*)(rowp + c * 16 + hi * 8);       // K group 0 (8 f16)
        a.q[1] = *(const int4*)(rowp + c * 16 + hi * 8 + 4);   // K group 1 (8 f16)
        union { int4 q[2]; v16h h; } b0, b1;
        b0.q[0] = wf[((size_t)(c)     * 32 + lane) * 2 + 0];
        b0.q[1] = wf[((size_t)(c)     * 32 + lane) * 2 + 1];
        b1.q[0] = wf[((size_t)(4 + c) * 32 + lane) * 2 + 0];
        b1.q[1] = wf[((size_t)(4 + c) * 32 + lane) * 2 + 1];
        acc0 = __builtin_amdgcn_wmma_f32_16x16x32_f16(false, a.h, false, b0.h,
                                                      (short)0, acc0, false, false);
        acc1 = __builtin_amdgcn_wmma_f32_16x16x32_f16(false, a.h, false, b1.h,
                                                      (short)0, acc1, false, false);
    }

    // ---- Phase 3: bias + sigmoid -> LDS sheaf tile (16 x 32) ---------------
    // C/D layout: lane holds (M = j + hi*8, N = lane&15) in acc[j].
    // sigmoid via native v_exp_f32 + single v_rcp_f32 (no precise-div chain).
    // biasPad is zero-padded to 32 entries: both loads unconditional, no EXEC
    // divergence before/after the WMMA block.
    float* sh = sheaf[wave];
    const float bn0 = biasPad[m];
    const float bn1 = biasPad[16 + m];
#pragma unroll
    for (int j = 0; j < 8; ++j) {
        int mm = j + hi * 8;
        float s0 = __builtin_amdgcn_rcpf(1.0f + __expf(-(acc0[j] + bn0)));
        sh[mm * 32 + m]       = s0;
        float s1 = __builtin_amdgcn_rcpf(1.0f + __expf(-(acc1[j] + bn1)));
        sh[mm * 32 + 16 + m]  = s1;            // cols 30,31 written but unused
    }
    asm volatile("s_wait_dscnt 0" ::: "memory");

    // ---- Phase 4: M[d][e] = sum_r A[d][r]*B[e][r] + C[d]*delta(d,e) --------
    // 16 rows x 36 outputs = 576 = 18 iterations x 32 lanes; contiguous stores.
    // (r, de) tracked incrementally: no integer division by 36 needed.
    int r  = 0;
    int de = lane;                 // lane < 32 < 36, so r starts at 0
    float* ob = out + (size_t)base * 36;
    const bool full = (base + 16 <= nnz);
#pragma unroll
    for (int it = 0; it < 18; ++it) {
        int d  = de / 6;           // 0..5 (cheap)
        int ee = de - d * 6;
        const float* srow = sh + r * 32;
        float a0 = srow[d * 2],        a1 = srow[d * 2 + 1];
        float b0 = srow[12 + ee * 2],  b1 = srow[12 + ee * 2 + 1];
        float mv = a0 * b0 + a1 * b1;
        if (d == ee) mv += srow[24 + d];
        if (full || (base + r) < nnz) ob[(size_t)r * 36 + de] = mv;
        de += 32;
        if (de >= 36) { de -= 36; ++r; }
    }
}

// ---------------------------------------------------------------------------
extern "C" void kernel_launch(void* const* d_in, const int* in_sizes, int n_in,
                              void* d_out, int out_size, void* d_ws, size_t ws_size,
                              hipStream_t stream)
{
    const float* x  = (const float*)d_in[0];   // (N*6, 64) f32
    const float* e  = (const float*)d_in[1];   // (E*6, 64) f32
    const float* W  = (const float*)d_in[2];   // (128, 30) f32
    const float* b  = (const float*)d_in[3];   // (30,) f32
    const int*   he = (const int*)d_in[4];     // (2, NNZ) i32
    // d_in[5] node_types, d_in[6] hyperedge_types: unused by the reference math

    const int N   = in_sizes[0] / (DD * FF);
    const int E   = in_sizes[1] / (DD * FF);
    const int NNZ = in_sizes[4] / 2;
    const int* rowi = he;
    const int* coli = he + NNZ;

    // workspace layout:
    //   [ W fragments 8192 B | biasPad 128 B (+pad to 256) | xm_h N*64 f16 | em_h E*64 f16 ]
    char* ws = (char*)d_ws;
    _Float16* wfrag   = (_Float16*)ws;                       // 8*32*16*2 = 8192 B
    float*    biasPad = (float*)(ws + 8192);                 // 32 floats
    _Float16* xmh     = (_Float16*)(ws + 8192 + 256);
    _Float16* emh     = xmh + (size_t)N * FF;

    wprep_kernel<<<1, 256, 0, stream>>>(W, b, wfrag, biasPad);

    long totalMean = (long)(N + E) * FF;
    int  mblocks   = (int)((totalMean + 255) / 256);
    means_kernel<<<mblocks, 256, 0, stream>>>(x, e, xmh, emh, N, E);

    int ntiles = (NNZ + 15) / 16;
    int blocks = (ntiles + 3) / 4;
    sheaf_main_kernel<<<blocks, 128, 0, stream>>>(xmh, emh, rowi, coli, wfrag, biasPad,
                                                  (float*)d_out, NNZ, ntiles);
}
